// HGT_86131274154722
// MI455X (gfx1250) — matrix operand order repaired and checked by reading, not compiled, not run on hardware
//
#include <hip/hip_runtime.h>
#include <math.h>

#define NA_N 100000
#define NP_N 200000
#define NE_E 500000
#define FA_F 256
#define FP_F 512
#define HC_C 128
#define NH_H 4
#define HD_D 32
#define NL_L 2
#define ATT_SCALE 0.17677669529663687f  // 1/sqrt(32)

typedef __attribute__((ext_vector_type(16))) _Float16 v16h;
typedef __attribute__((ext_vector_type(8)))  _Float16 v8h;
typedef __attribute__((ext_vector_type(8)))  float    v8f;

// ---------------------------------------------------------------- utilities

__device__ __forceinline__ float atomicMaxFloat(float* addr, float value) {
  int* ia = (int*)addr;
  int old = __float_as_int(*addr);
  while (__int_as_float(old) < value) {
    int assumed = old;
    old = atomicCAS(ia, assumed, __float_as_int(value));
    if (old == assumed) break;
  }
  return __int_as_float(old);
}

__global__ void k_fill_f32(float* __restrict__ p, float v, long long n) {
  long long i = (long long)blockIdx.x * blockDim.x + threadIdx.x;
  if (i < n) p[i] = v;
}

__global__ void k_cvt_f16(const float* __restrict__ x, _Float16* __restrict__ y, long long n) {
  long long i = (long long)blockIdx.x * blockDim.x + threadIdx.x;
  if (i < n) y[i] = (_Float16)x[i];
}

// exact GeLU (approximate=False) fused into the f16 down-convert
__global__ void k_cvt_f16_gelu(const float* __restrict__ x, _Float16* __restrict__ y, long long n) {
  long long i = (long long)blockIdx.x * blockDim.x + threadIdx.x;
  if (i < n) {
    float v = x[i];
    float g = 0.5f * v * (1.0f + erff(v * 0.70710678118654752f));
    y[i] = (_Float16)g;
  }
}

// Bt[n*K + k] = (f16) W[k*N + n]   (W is K x N row-major)
__global__ void k_transpose_cvt(const float* __restrict__ W, _Float16* __restrict__ Bt,
                                int K, int N) {
  long long i = (long long)blockIdx.x * blockDim.x + threadIdx.x;
  if (i >= (long long)K * N) return;
  int k = (int)(i % K);
  int n = (int)(i / K);
  Bt[i] = (_Float16)W[(long long)k * N + n];
}

// ---------------------------------------------------------------- WMMA GEMM
// C[M,N](f32) = A[M,K](f16) @ B[K,N] + bias[N], B given transposed: Bt[N,K](f16)
// act: 0 = none, 1 = relu.  M%32==0, N%128==0, K%32==0 (guaranteed by problem dims).
// block = 128 threads = 4 waves laid out 2(M) x 2(N); each wave owns a 16x64
// output strip: 4 f32 accumulators sharing one A fragment per K-step, with the
// next K-step's fragments preloaded so loads overlap the 4-wmma burst.
#define WMMA_F16(a, b, c) \
  __builtin_amdgcn_wmma_f32_16x16x32_f16(false, (a), false, (b), (short)0, (c), false, false)

__global__ __launch_bounds__(128) void k_gemm_f16(
    const _Float16* __restrict__ A, const _Float16* __restrict__ Bt,
    const float* __restrict__ bias, float* __restrict__ C,
    int M, int N, int K, int act)
{
  const int lane = threadIdx.x & 31;
  const int wave = threadIdx.x >> 5;
  const int tileM = blockIdx.x * 32 + (wave >> 1) * 16;
  const int tileN = blockIdx.y * 128 + (wave & 1) * 64;

  const int m  = tileM + (lane & 15);   // A fragment row (lanes 16-31 mirror 0-15)
  const int nf = tileN + (lane & 15);   // B fragment base column
  const int kbA = (lane >> 4) << 4;     // A: upper lanes hold K+16 .. (matches B)
  const int kbB = (lane >> 4) << 4;     // B: upper lanes hold K+16 .. (contiguous k order, same as A)

  const _Float16* __restrict__ arow  = A  + (long long)m * K + kbA;
  const _Float16* __restrict__ brow0 = Bt + (long long)(nf +  0) * K + kbB;
  const _Float16* __restrict__ brow1 = Bt + (long long)(nf + 16) * K + kbB;
  const _Float16* __restrict__ brow2 = Bt + (long long)(nf + 32) * K + kbB;
  const _Float16* __restrict__ brow3 = Bt + (long long)(nf + 48) * K + kbB;

  auto loadA = [&](int k0) -> v16h {
    return *(const v16h*)(arow + k0);         // K = k0+kbA .. +15
  };

  v8f acc0 = {}, acc1 = {}, acc2 = {}, acc3 = {};

  v16h a  = loadA(0);
  v16h b0 = *(const v16h*)(brow0);
  v16h b1 = *(const v16h*)(brow1);
  v16h b2 = *(const v16h*)(brow2);
  v16h b3 = *(const v16h*)(brow3);

  for (int k0 = 0; k0 < K; k0 += 32) {
    const int kn = k0 + 32;
    v16h an = a, bn0 = b0, bn1 = b1, bn2 = b2, bn3 = b3;
    if (kn < K) {
      __builtin_prefetch(arow + kn + 32, 0, 3);   // gfx1250 global_prefetch_b8
      an  = loadA(kn);
      bn0 = *(const v16h*)(brow0 + kn);
      bn1 = *(const v16h*)(brow1 + kn);
      bn2 = *(const v16h*)(brow2 + kn);
      bn3 = *(const v16h*)(brow3 + kn);
    }
    acc0 = WMMA_F16(a, b0, acc0);
    acc1 = WMMA_F16(a, b1, acc1);
    acc2 = WMMA_F16(a, b2, acc2);
    acc3 = WMMA_F16(a, b3, acc3);
    a = an; b0 = bn0; b1 = bn1; b2 = bn2; b3 = bn3;
  }

  const int rowBase = tileM + ((lane >> 4) << 3);  // lanes>=16 -> rows +8
  auto store = [&](v8f acc, int col) {
    const float bv = bias[col];
    #pragma unroll
    for (int r = 0; r < 8; ++r) {
      float v = acc[r] + bv;
      if (act == 1) v = fmaxf(v, 0.0f);
      C[(long long)(rowBase + r) * N + col] = v;
    }
  };
  store(acc0, nf + 0);
  store(acc1, nf + 16);
  store(acc2, nf + 32);
  store(acc3, nf + 48);
}

// ------------------------------------------------- per-head 32x32 transforms
// out[n, h*32+e] = sum_d kqv[n, slot*128 + h*32 + d] * w[h, d, e]
__global__ __launch_bounds__(256) void k_head_transform(
    const float* __restrict__ kqv, const float* __restrict__ w,
    float* __restrict__ out, int n_nodes, int slot)
{
  __shared__ float sw[NH_H * HD_D * HD_D];           // 16 KB
  for (int i = threadIdx.x; i < NH_H * HD_D * HD_D; i += 256) sw[i] = w[i];
  __syncthreads();
  long long idx = (long long)blockIdx.x * 256 + threadIdx.x;
  if (idx >= (long long)n_nodes * HC_C) return;
  int c = (int)(idx & (HC_C - 1));
  long long nn = idx >> 7;
  int h = c >> 5, e = c & (HD_D - 1);
  const float* src = kqv + nn * (3 * HC_C) + slot * HC_C + h * HD_D;
  const float* wh  = sw + h * HD_D * HD_D + e;
  float s = 0.0f;
  #pragma unroll
  for (int d = 0; d < HD_D; ++d) s += src[d] * wh[d * HD_D];
  out[idx] = s;
}

// ---------------------------------------------------------------- edge phase
// one wave per edge; lane covers 4 contiguous channels; 8 lanes per head.
__global__ __launch_bounds__(256) void k_edge_alpha(
    const float* __restrict__ kqv_dst, const float* __restrict__ kp,
    const float* __restrict__ prel, const int* __restrict__ src,
    const int* __restrict__ dst, float* __restrict__ alpha,
    float* __restrict__ mmax, int nE)
{
  const int lane = threadIdx.x & 31;
  const int e = blockIdx.x * 8 + (threadIdx.x >> 5);
  if (e >= nE) return;
  const int s = src[e], d = dst[e];
  const float4 q = *(const float4*)(kqv_dst + (long long)d * 384 + 128 + lane * 4); // slot 1 = Q
  const float4 k = *(const float4*)(kp + (long long)s * 128 + lane * 4);
  float part = q.x * k.x + q.y * k.y + q.z * k.z + q.w * k.w;
  part += __shfl_xor(part, 1, 32);
  part += __shfl_xor(part, 2, 32);
  part += __shfl_xor(part, 4, 32);
  const int h = lane >> 3;
  if ((lane & 7) == 0) {
    float a = part * prel[h] * ATT_SCALE;
    alpha[(long long)e * NH_H + h] = a;
    atomicMaxFloat(&mmax[(long long)d * NH_H + h], a);
  }
}

__global__ void k_edge_expsum(float* __restrict__ alpha, const int* __restrict__ dst,
                              const float* __restrict__ mmax, float* __restrict__ den,
                              long long n)
{
  long long i = (long long)blockIdx.x * blockDim.x + threadIdx.x;
  if (i >= n) return;
  int e = (int)(i >> 2), h = (int)(i & 3);
  int d = dst[e];
  float ex = expf(alpha[i] - mmax[(long long)d * NH_H + h]);
  alpha[i] = ex;
  atomicAdd(&den[(long long)d * NH_H + h], ex);
}

__global__ __launch_bounds__(256) void k_edge_scatter(
    const float* __restrict__ vp, const float* __restrict__ alpha,
    const float* __restrict__ den, const int* __restrict__ src,
    const int* __restrict__ dst, float* __restrict__ agg, int nE)
{
  const int lane = threadIdx.x & 31;
  const int e = blockIdx.x * 8 + (threadIdx.x >> 5);
  if (e >= nE) return;
  const int s = src[e], d = dst[e];
  const int h = lane >> 3;
  const float a = alpha[(long long)e * NH_H + h] / (den[(long long)d * NH_H + h] + 1e-16f);
  const float4 v = *(const float4*)(vp + (long long)s * 128 + lane * 4);
  float* out = agg + (long long)d * 128 + lane * 4;
  atomicAdd(out + 0, v.x * a);
  atomicAdd(out + 1, v.y * a);
  atomicAdd(out + 2, v.z * a);
  atomicAdd(out + 3, v.w * a);
}

// x_new = sigmoid(skip)*o + (1-sigmoid(skip))*x_old
__global__ void k_blend(const float* __restrict__ o, const float* __restrict__ xold,
                        const float* __restrict__ skip, float* __restrict__ xnew,
                        long long n)
{
  long long i = (long long)blockIdx.x * blockDim.x + threadIdx.x;
  if (i >= n) return;
  float b = 1.0f / (1.0f + expf(-skip[0]));
  xnew[i] = b * o[i] + (1.0f - b) * xold[i];
}

// ---------------------------------------------------------------- launcher

extern "C" void kernel_launch(void* const* d_in, const int* in_sizes, int n_in,
                              void* d_out, int out_size, void* d_ws, size_t ws_size,
                              hipStream_t stream)
{
  const float* x_author = (const float*)d_in[0];
  const float* x_paper  = (const float*)d_in[1];
  const float* lin_a_w  = (const float*)d_in[2];
  const float* lin_a_b  = (const float*)d_in[3];
  const float* lin_p_w  = (const float*)d_in[4];
  const float* lin_p_b  = (const float*)d_in[5];
  const float* kqv_w    = (const float*)d_in[6];
  const float* kqv_b    = (const float*)d_in[7];
  const float* out_w    = (const float*)d_in[8];
  const float* out_b    = (const float*)d_in[9];
  const float* skip     = (const float*)d_in[10];
  const float* wk_rel   = (const float*)d_in[11];
  const float* wv_rel   = (const float*)d_in[12];
  const float* p_rel    = (const float*)d_in[13];
  const int* src_pa = (const int*)d_in[14];
  const int* dst_pa = (const int*)d_in[15];
  const int* src_ap = (const int*)d_in[16];
  const int* dst_ap = (const int*)d_in[17];

  float* out_xa = (float*)d_out;
  float* out_xp = (float*)d_out + (long long)NA_N * HC_C;

  char* wsb = (char*)d_ws;
  size_t off = 0;
  auto alloc = [&](size_t bytes) -> void* {
    off = (off + 255) & ~(size_t)255;
    void* p = wsb + off;
    off += bytes;
    return p;
  };

  // f16 transposed weights
  _Float16* t_lin_a = (_Float16*)alloc((size_t)HC_C * FA_F * 2);
  _Float16* t_lin_p = (_Float16*)alloc((size_t)HC_C * FP_F * 2);
  _Float16* t_kqv   = (_Float16*)alloc((size_t)NL_L * 2 * 3 * HC_C * HC_C * 2);
  _Float16* t_out   = (_Float16*)alloc((size_t)NL_L * 2 * HC_C * HC_C * 2);
  // f16 activations
  _Float16* hxa_big = (_Float16*)alloc((size_t)NA_N * FA_F * 2);
  _Float16* hxp_big = (_Float16*)alloc((size_t)NP_N * FP_F * 2);
  _Float16* hxa     = (_Float16*)alloc((size_t)NA_N * HC_C * 2);
  _Float16* hxp     = (_Float16*)alloc((size_t)NP_N * HC_C * 2);
  // f32 state
  float* xa[2] = { (float*)alloc((size_t)NA_N * HC_C * 4), (float*)alloc((size_t)NA_N * HC_C * 4) };
  float* xp[2] = { (float*)alloc((size_t)NP_N * HC_C * 4), (float*)alloc((size_t)NP_N * HC_C * 4) };
  float* kqva = (float*)alloc((size_t)NA_N * 3 * HC_C * 4);
  float* kqvp = (float*)alloc((size_t)NP_N * 3 * HC_C * 4);
  float* kp_p = (float*)alloc((size_t)NP_N * HC_C * 4);
  float* vp_p = (float*)alloc((size_t)NP_N * HC_C * 4);
  float* kp_a = (float*)alloc((size_t)NA_N * HC_C * 4);
  float* vp_a = (float*)alloc((size_t)NA_N * HC_C * 4);
  float* alpha = (float*)alloc((size_t)NE_E * NH_H * 4);
  float* mmax  = (float*)alloc((size_t)NP_N * NH_H * 4);
  float* den   = (float*)alloc((size_t)NP_N * NH_H * 4);
  float* agg_a = (float*)alloc((size_t)NA_N * HC_C * 4);
  float* agg_p = (float*)alloc((size_t)NP_N * HC_C * 4);
  float* o_a = kqva;   // kqv buffers dead by the time o is produced
  float* o_p = kqvp;

  auto nblk = [](long long n) { return (unsigned)((n + 255) / 256); };
  auto cvt = [&](const float* x, _Float16* y, long long n) {
    k_cvt_f16<<<nblk(n), 256, 0, stream>>>(x, y, n);
  };
  auto gelu_cvt = [&](const float* x, _Float16* y, long long n) {
    k_cvt_f16_gelu<<<nblk(n), 256, 0, stream>>>(x, y, n);
  };
  auto fill = [&](float* p, float v, long long n) {
    k_fill_f32<<<nblk(n), 256, 0, stream>>>(p, v, n);
  };
  auto gemm = [&](const _Float16* A, const _Float16* Bt, const float* bias,
                  float* C, int M, int N, int K, int act) {
    dim3 grid(M / 32, N / 128);
    k_gemm_f16<<<grid, 128, 0, stream>>>(A, Bt, bias, C, M, N, K, act);
  };
  auto tcvt = [&](const float* W, _Float16* Bt, int K, int N) {
    k_transpose_cvt<<<nblk((long long)K * N), 256, 0, stream>>>(W, Bt, K, N);
  };

  // ---- weight prep (once) ----
  tcvt(lin_a_w, t_lin_a, FA_F, HC_C);
  tcvt(lin_p_w, t_lin_p, FP_F, HC_C);
  for (int i = 0; i < NL_L * 2; ++i) {
    tcvt(kqv_w + (long long)i * HC_C * 3 * HC_C, t_kqv + (long long)i * 3 * HC_C * HC_C, HC_C, 3 * HC_C);
    tcvt(out_w + (long long)i * HC_C * HC_C,     t_out + (long long)i * HC_C * HC_C,     HC_C, HC_C);
  }

  // ---- input projections (relu) ----
  cvt(x_author, hxa_big, (long long)NA_N * FA_F);
  cvt(x_paper,  hxp_big, (long long)NP_N * FP_F);
  gemm(hxa_big, t_lin_a, lin_a_b, xa[0], NA_N, HC_C, FA_F, 1);
  gemm(hxp_big, t_lin_p, lin_p_b, xp[0], NP_N, HC_C, FP_F, 1);

  int cur = 0;
  for (int l = 0; l < NL_L; ++l) {
    const int ia = l * 2 + 0, ip = l * 2 + 1;
    float* cxa = xa[cur];
    float* cxp = xp[cur];
    float* nxa = (l == NL_L - 1) ? out_xa : xa[1 - cur];
    float* nxp = (l == NL_L - 1) ? out_xp : xp[1 - cur];

    // kqv projections
    cvt(cxa, hxa, (long long)NA_N * HC_C);
    cvt(cxp, hxp, (long long)NP_N * HC_C);
    gemm(hxa, t_kqv + (long long)ia * 3 * HC_C * HC_C, kqv_b + (long long)ia * 3 * HC_C,
         kqva, NA_N, 3 * HC_C, HC_C, 0);
    gemm(hxp, t_kqv + (long long)ip * 3 * HC_C * HC_C, kqv_b + (long long)ip * 3 * HC_C,
         kqvp, NP_N, 3 * HC_C, HC_C, 0);

    // per-head k'/v' (edge-type 0: papers feed authors; edge-type 1: authors feed papers)
    const unsigned nbp = nblk((long long)NP_N * HC_C);
    const unsigned nba = nblk((long long)NA_N * HC_C);
    k_head_transform<<<nbp, 256, 0, stream>>>(kqvp, wk_rel + (long long)ia * NH_H * HD_D * HD_D, kp_p, NP_N, 0);
    k_head_transform<<<nbp, 256, 0, stream>>>(kqvp, wv_rel + (long long)ia * NH_H * HD_D * HD_D, vp_p, NP_N, 2);
    k_head_transform<<<nba, 256, 0, stream>>>(kqva, wk_rel + (long long)ip * NH_H * HD_D * HD_D, kp_a, NA_N, 0);
    k_head_transform<<<nba, 256, 0, stream>>>(kqva, wv_rel + (long long)ip * NH_H * HD_D * HD_D, vp_a, NA_N, 2);

    // paper -> author edges
    fill(mmax, -INFINITY, (long long)NA_N * NH_H);
    fill(den, 0.0f, (long long)NA_N * NH_H);
    fill(agg_a, 0.0f, (long long)NA_N * HC_C);
    k_edge_alpha<<<(NE_E + 7) / 8, 256, 0, stream>>>(kqva, kp_p, p_rel + ia * NH_H,
                                                     src_pa, dst_pa, alpha, mmax, NE_E);
    k_edge_expsum<<<nblk((long long)NE_E * NH_H), 256, 0, stream>>>(alpha, dst_pa, mmax, den,
                                                                    (long long)NE_E * NH_H);
    k_edge_scatter<<<(NE_E + 7) / 8, 256, 0, stream>>>(vp_p, alpha, den, src_pa, dst_pa, agg_a, NE_E);

    // author -> paper edges
    fill(mmax, -INFINITY, (long long)NP_N * NH_H);
    fill(den, 0.0f, (long long)NP_N * NH_H);
    fill(agg_p, 0.0f, (long long)NP_N * HC_C);
    k_edge_alpha<<<(NE_E + 7) / 8, 256, 0, stream>>>(kqvp, kp_a, p_rel + ip * NH_H,
                                                     src_ap, dst_ap, alpha, mmax, NE_E);
    k_edge_expsum<<<nblk((long long)NE_E * NH_H), 256, 0, stream>>>(alpha, dst_ap, mmax, den,
                                                                    (long long)NE_E * NH_H);
    k_edge_scatter<<<(NE_E + 7) / 8, 256, 0, stream>>>(vp_a, alpha, den, src_ap, dst_ap, agg_p, NE_E);

    // output projections + skip blend
    gelu_cvt(agg_a, hxa, (long long)NA_N * HC_C);
    gelu_cvt(agg_p, hxp, (long long)NP_N * HC_C);
    gemm(hxa, t_out + (long long)ia * HC_C * HC_C, out_b + (long long)ia * HC_C, o_a, NA_N, HC_C, HC_C, 0);
    gemm(hxp, t_out + (long long)ip * HC_C * HC_C, out_b + (long long)ip * HC_C, o_p, NP_N, HC_C, HC_C, 0);
    k_blend<<<nblk((long long)NA_N * HC_C), 256, 0, stream>>>(o_a, cxa, skip + ia, nxa, (long long)NA_N * HC_C);
    k_blend<<<nblk((long long)NP_N * HC_C), 256, 0, stream>>>(o_p, cxp, skip + ip, nxp, (long long)NP_N * HC_C);

    cur ^= 1;
  }
}
